// TransformerDecoder_24369644437648
// MI455X (gfx1250) — matrix operand order repaired and checked by reading, not compiled, not run on hardware
//
#include <hip/hip_runtime.h>
#include <hip/hip_bf16.h>

// ---------------------------------------------------------------------------
// GPT-style decoder forward (B=2, T=2048, D=512, H=8, L=6, V=32000) for
// gfx1250 (MI455X). All matmuls via v_wmma_f32_16x16x32_bf16 (wave32).
// ---------------------------------------------------------------------------

typedef __bf16 bf16_t;
typedef __attribute__((ext_vector_type(16))) __bf16 bf16x16;
typedef __attribute__((ext_vector_type(8)))  __bf16 bf16x8;
typedef __attribute__((ext_vector_type(8)))  float  f32x8;

#define DV   512
#define HV   8
#define LV   6
#define TV   2048
#define BV   2
#define VV   32000
#define MV   (BV * TV)   /* 4096 rows */
#define D4V  (4 * DV)    /* 2048 */

static __device__ __forceinline__ f32x8 zero8() {
  f32x8 z = {0.f, 0.f, 0.f, 0.f, 0.f, 0.f, 0.f, 0.f};
  return z;
}

static __device__ __forceinline__ f32x8 wmma_bf16(bf16x16 a, bf16x16 b, f32x8 c) {
  // 8 args: (neg_a, A, neg_b, B, c_mod, C, reuse_a, reuse_b)
  return __builtin_amdgcn_wmma_f32_16x16x32_bf16(false, a, false, b, (short)0, c,
                                                 false, false);
}

// A-matrix fragment, 16x32 bf16, source row-major with leading dim `ld`.
// ISA layout: lanes 0-15 -> row = lane, K in {0..7, 16..23};
//             lanes 16-31 -> row = lane-16, K in {8..15, 24..31}.
static __device__ __forceinline__ bf16x16 load_frag_a(const bf16_t* base, int ld,
                                                      int lane) {
  int row = lane & 15;
  int k0  = (lane & 16) ? 8 : 0;
  union { bf16x16 v; bf16x8 h[2]; } u;
  const bf16_t* p = base + (size_t)row * ld;
  u.h[0] = *(const bf16x8*)(p + k0);
  u.h[1] = *(const bf16x8*)(p + k0 + 16);
  return u.v;
}

// B-matrix fragment, 32x16 bf16. Element (k,n) is read from bt[n*ldb + k]
// (i.e. the "N-major / K-contiguous" layout). ISA layout: lanes 0-15 ->
// N = lane, K in {0..7, 16..23}; lanes 16-31 -> N = lane-16, K in {8..15, 24..31}.
static __device__ __forceinline__ bf16x16 load_frag_b(const bf16_t* bt, int ldb,
                                                      int lane) {
  int n  = lane & 15;
  int k0 = (lane & 16) ? 8 : 0;
  union { bf16x16 v; bf16x8 h[2]; } u;
  const bf16_t* p = bt + (size_t)n * ldb + k0;
  u.h[0] = *(const bf16x8*)(p);
  u.h[1] = *(const bf16x8*)(p + 16);
  return u.v;
}

// ---------------------------------------------------------------------------
// Weight convert + transpose: fp32 W[K,N] -> bf16 Wt[N,K]
// ---------------------------------------------------------------------------
__global__ __launch_bounds__(256) void wconv_kernel(const float* __restrict__ w,
                                                    bf16_t* __restrict__ wt,
                                                    int K, int N) {
  long i = (long)blockIdx.x * 256 + threadIdx.x;  // index into output [N,K]
  if (i >= (long)K * N) return;
  int  k = (int)(i % K);
  long n = i / K;
  wt[i] = (bf16_t)w[(size_t)k * N + n];
}

// ---------------------------------------------------------------------------
// Embedding: x[b,t,:] = tok_emb[idx[b,t],:] + pos_emb[t,:]
// ---------------------------------------------------------------------------
__global__ __launch_bounds__(256) void embed_kernel(const int* __restrict__ idx,
                                                    const float* __restrict__ tok,
                                                    const float* __restrict__ pos,
                                                    float* __restrict__ x) {
  int i   = blockIdx.x * 256 + threadIdx.x;  // over MV*DV
  int row = i >> 9;                          // D = 512
  int d   = i & 511;
  int t   = row & (TV - 1);
  x[i] = tok[(size_t)idx[row] * DV + d] + pos[(size_t)t * DV + d];
}

// ---------------------------------------------------------------------------
// LayerNorm row kernel: fp32 in, bf16 out (GEMM input precision).
// ---------------------------------------------------------------------------
__global__ __launch_bounds__(128) void ln_kernel(const float* __restrict__ x,
                                                 const float* __restrict__ g,
                                                 const float* __restrict__ be,
                                                 bf16_t* __restrict__ out, int D) {
  int row = blockIdx.x;
  int tid = threadIdx.x;
  const float* xr = x + (size_t)row * D;
  __shared__ float red[128];

  float s = 0.f;
  for (int d = tid; d < D; d += 128) s += xr[d];
  red[tid] = s;
  __syncthreads();
  for (int o = 64; o > 0; o >>= 1) {
    if (tid < o) red[tid] += red[tid + o];
    __syncthreads();
  }
  float mean = red[0] / (float)D;
  __syncthreads();

  float v = 0.f;
  for (int d = tid; d < D; d += 128) {
    float t = xr[d] - mean;
    v += t * t;
  }
  red[tid] = v;
  __syncthreads();
  for (int o = 64; o > 0; o >>= 1) {
    if (tid < o) red[tid] += red[tid + o];
    __syncthreads();
  }
  float rstd = rsqrtf(red[0] / (float)D + 1e-5f);

  for (int d = tid; d < D; d += 128)
    out[(size_t)row * D + d] = (bf16_t)((xr[d] - mean) * rstd * g[d] + be[d]);
}

// ---------------------------------------------------------------------------
// Generic WMMA GEMM: C[M,N] = A[M,K] (bf16, row-major) x Bt[N,K] (bf16, W^T)
// Epilogue: optional bias, relu, fp32 residual add; fp32 and/or bf16 output.
// Block = 256 threads (8 waves); wave tile 32x32; block tile 64x128.
// ---------------------------------------------------------------------------
__global__ __launch_bounds__(256) void gemm_kernel(const bf16_t* __restrict__ A,
                                                   const bf16_t* __restrict__ Bt,
                                                   int M, int N, int K,
                                                   const float* __restrict__ bias,
                                                   const float* __restrict__ resid,
                                                   float* __restrict__ outF,
                                                   bf16_t* __restrict__ outB,
                                                   int relu) {
  int lane = threadIdx.x & 31;
  int warp = threadIdx.x >> 5;
  int m0 = blockIdx.x * 64 + (warp >> 2) * 32;
  int n0 = blockIdx.y * 128 + (warp & 3) * 32;

  f32x8 cc[2][2];
  cc[0][0] = zero8(); cc[0][1] = zero8();
  cc[1][0] = zero8(); cc[1][1] = zero8();

  for (int k0 = 0; k0 < K; k0 += 32) {
    if (k0 + 32 < K) {  // stream the next K-slab through L2
      __builtin_prefetch(A + (size_t)m0 * K + k0 + 32, 0, 1);
      __builtin_prefetch(Bt + (size_t)n0 * K + k0 + 32, 0, 1);
    }
    bf16x16 a0 = load_frag_a(A + (size_t)(m0)      * K + k0, K, lane);
    bf16x16 a1 = load_frag_a(A + (size_t)(m0 + 16) * K + k0, K, lane);
    bf16x16 b0 = load_frag_b(Bt + (size_t)(n0)      * K + k0, K, lane);
    bf16x16 b1 = load_frag_b(Bt + (size_t)(n0 + 16) * K + k0, K, lane);
    cc[0][0] = wmma_bf16(a0, b0, cc[0][0]);
    cc[0][1] = wmma_bf16(a0, b1, cc[0][1]);
    cc[1][0] = wmma_bf16(a1, b0, cc[1][0]);
    cc[1][1] = wmma_bf16(a1, b1, cc[1][1]);
  }

  int half8 = (lane & 16) ? 8 : 0;
  int n     = lane & 15;
  for (int i = 0; i < 2; ++i)
    for (int j = 0; j < 2; ++j)
      for (int r = 0; r < 8; ++r) {
        int row = m0 + 16 * i + half8 + r;
        int col = n0 + 16 * j + n;
        size_t o = (size_t)row * N + col;
        float v = cc[i][j][r];
        if (bias)  v += bias[col];
        if (relu)  v = fmaxf(v, 0.f);
        if (resid) v += resid[o];
        if (outF)  outF[o] = v;
        if (outB)  outB[o] = (bf16_t)v;
      }
}

// ---------------------------------------------------------------------------
// V transpose: vb [B,T,D] bf16 -> vt [B,H,hd,T] bf16 (so P x V B-fragments
// are K-contiguous along the key axis).
// ---------------------------------------------------------------------------
__global__ __launch_bounds__(256) void vtr_kernel(const bf16_t* __restrict__ vb,
                                                  bf16_t* __restrict__ vt) {
  int i    = blockIdx.x * 256 + threadIdx.x;  // over MV*DV, output index
  int t    = i & (TV - 1);
  int rest = i >> 11;
  int e    = rest & 63;
  int h    = (rest >> 6) & (HV - 1);
  int b    = rest >> 9;
  vt[i] = vb[((size_t)(b * TV + t)) * DV + h * 64 + e];
}

// ---------------------------------------------------------------------------
// Flash-style causal attention, one 16-query tile per wave.
// S = Q K^T (WMMA) -> online softmax -> O += P V (WMMA via LDS re-swizzle).
// ---------------------------------------------------------------------------
__global__ __launch_bounds__(256) void attn_kernel(const bf16_t* __restrict__ qb,
                                                   const bf16_t* __restrict__ kb,
                                                   const bf16_t* __restrict__ vt,
                                                   bf16_t* __restrict__ ob,
                                                   float scale) {
  __shared__ __align__(16) bf16_t plds[8][16 * 32];  // 1 KB per wave
  int lane = threadIdx.x & 31;
  int warp = threadIdx.x >> 5;
  int wid  = blockIdx.x * 8 + warp;      // B*H*(T/16) = 2048 waves
  int qt   = wid & 127;                  // T/16 = 128
  int h    = (wid >> 7) & (HV - 1);
  int b    = wid >> 10;
  int bT   = b * TV;

  // Q fragments are loop-invariant: 16 queries x 64 head-dims.
  const bf16_t* qp = qb + (size_t)(bT + qt * 16) * DV + h * 64;
  bf16x16 qa0 = load_frag_a(qp, DV, lane);       // hd 0..31
  bf16x16 qa1 = load_frag_a(qp + 32, DV, lane);  // hd 32..63

  f32x8 acc[4];
  float mi[8], li[8];
  for (int c = 0; c < 4; ++c) acc[c] = zero8();
  for (int r = 0; r < 8; ++r) { mi[r] = -1e30f; li[r] = 0.f; }

  int half8 = (lane & 16) ? 8 : 0;
  int n     = lane & 15;
  bf16_t* pl = &plds[warp][0];

  int ktEnd = (qt * 16 + 15) >> 5;  // last 32-key tile touching this q-tile
  for (int kt = 0; kt <= ktEnd; ++kt) {
    // ---- S = Q K^T for 32 keys (two 16-key column halves) -----------------
    f32x8 s[2];
    for (int cH = 0; cH < 2; ++cH) {
      const bf16_t* kp = kb + (size_t)(bT + kt * 32 + cH * 16) * DV + h * 64;
      bf16x16 kb0 = load_frag_b(kp, DV, lane);       // hd 0..31
      bf16x16 kb1 = load_frag_b(kp + 32, DV, lane);  // hd 32..63
      f32x8 sc = zero8();
      sc = wmma_bf16(qa0, kb0, sc);
      sc = wmma_bf16(qa1, kb1, sc);
      s[cH] = sc;
    }
    // ---- scale + causal mask + row max (half-wave shfl reductions) --------
    float tm[8];
    for (int r = 0; r < 8; ++r) {
      int qg = qt * 16 + half8 + r;
      float s0 = s[0][r] * scale;
      float s1 = s[1][r] * scale;
      if (kt * 32 + n > qg)      s0 = -1e30f;
      if (kt * 32 + 16 + n > qg) s1 = -1e30f;
      s[0][r] = s0; s[1][r] = s1;
      float v = fmaxf(s0, s1);
      v = fmaxf(v, __shfl_xor(v, 1, 32));
      v = fmaxf(v, __shfl_xor(v, 2, 32));
      v = fmaxf(v, __shfl_xor(v, 4, 32));
      v = fmaxf(v, __shfl_xor(v, 8, 32));
      tm[r] = v;
    }
    // ---- online softmax update; stage P (bf16) into LDS -------------------
    for (int r = 0; r < 8; ++r) {
      float mnew = fmaxf(mi[r], tm[r]);
      float corr = __expf(mi[r] - mnew);
      float p0 = __expf(s[0][r] - mnew);
      float p1 = __expf(s[1][r] - mnew);
      float ps = p0 + p1;
      ps += __shfl_xor(ps, 1, 32);
      ps += __shfl_xor(ps, 2, 32);
      ps += __shfl_xor(ps, 4, 32);
      ps += __shfl_xor(ps, 8, 32);
      li[r] = li[r] * corr + ps;
      mi[r] = mnew;
      for (int c = 0; c < 4; ++c) acc[c][r] *= corr;
      int m = half8 + r;
      pl[m * 32 + n]      = (bf16_t)p0;   // C layout -> LDS row-major 16x32
      pl[m * 32 + 16 + n] = (bf16_t)p1;
    }
    asm volatile("s_wait_dscnt 0" ::: "memory");  // RAW through LDS (same wave)
    // ---- O += P V ----------------------------------------------------------
    bf16x16 pa = load_frag_a(pl, 32, lane);  // A-layout reload of P
    const bf16_t* vp = vt + ((size_t)(b * HV + h) * 64) * TV + kt * 32;
    for (int c = 0; c < 4; ++c) {
      bf16x16 vf = load_frag_b(vp + (size_t)(c * 16) * TV, TV, lane);
      acc[c] = wmma_bf16(pa, vf, acc[c]);
    }
  }
  // ---- epilogue: O /= l, write bf16 [B,T,D] --------------------------------
  for (int c = 0; c < 4; ++c)
    for (int r = 0; r < 8; ++r) {
      float o = acc[c][r] / li[r];
      int row = bT + qt * 16 + half8 + r;
      ob[(size_t)row * DV + h * 64 + c * 16 + n] = (bf16_t)o;
    }
}

// ---------------------------------------------------------------------------
// Per-row NLL: rl[row] = -(logit[tgt] - max - log(sum exp(logit - max)))
// ---------------------------------------------------------------------------
__global__ __launch_bounds__(256) void loss_row_kernel(const float* __restrict__ logits,
                                                       const int* __restrict__ tgt,
                                                       float* __restrict__ rl) {
  int row = blockIdx.x;
  int tid = threadIdx.x;
  const float* lr = logits + (size_t)row * VV;
  __shared__ float red[256];

  float m = -1e30f;
  for (int i = tid; i < VV; i += 256) m = fmaxf(m, lr[i]);
  red[tid] = m;
  __syncthreads();
  for (int o = 128; o > 0; o >>= 1) {
    if (tid < o) red[tid] = fmaxf(red[tid], red[tid + o]);
    __syncthreads();
  }
  float mx = red[0];
  __syncthreads();

  float s = 0.f;
  for (int i = tid; i < VV; i += 256) s += __expf(lr[i] - mx);
  red[tid] = s;
  __syncthreads();
  for (int o = 128; o > 0; o >>= 1) {
    if (tid < o) red[tid] += red[tid + o];
    __syncthreads();
  }
  if (tid == 0) rl[row] = -(lr[tgt[row]] - mx - __logf(red[0]));
}

__global__ __launch_bounds__(256) void loss_reduce_kernel(const float* __restrict__ rl,
                                                          float* __restrict__ out) {
  __shared__ float red[256];
  float s = 0.f;
  for (int i = threadIdx.x; i < MV; i += 256) s += rl[i];
  red[threadIdx.x] = s;
  __syncthreads();
  for (int o = 128; o > 0; o >>= 1) {
    if (threadIdx.x < o) red[threadIdx.x] += red[threadIdx.x + o];
    __syncthreads();
  }
  if (threadIdx.x == 0) out[0] = red[0] / (float)MV;
}

// ---------------------------------------------------------------------------
// Host orchestration
// ---------------------------------------------------------------------------
extern "C" void kernel_launch(void* const* d_in, const int* in_sizes, int n_in,
                              void* d_out, int out_size, void* d_ws, size_t ws_size,
                              hipStream_t stream) {
  (void)in_sizes; (void)n_in; (void)out_size; (void)ws_size;
  const int*   idx  = (const int*)d_in[0];
  const int*   tgt  = (const int*)d_in[1];
  const float* tok  = (const float*)d_in[2];
  const float* pos  = (const float*)d_in[3];
  const float* Wq   = (const float*)d_in[4];
  const float* Wk   = (const float*)d_in[5];
  const float* Wv   = (const float*)d_in[6];
  const float* Wo   = (const float*)d_in[7];
  const float* bo   = (const float*)d_in[8];
  const float* W1   = (const float*)d_in[9];
  const float* b1   = (const float*)d_in[10];
  const float* W2   = (const float*)d_in[11];
  const float* b2   = (const float*)d_in[12];
  const float* ln1g = (const float*)d_in[13];
  const float* ln1b = (const float*)d_in[14];
  const float* ln2g = (const float*)d_in[15];
  const float* ln2b = (const float*)d_in[16];
  const float* lnfg = (const float*)d_in[17];
  const float* lnfb = (const float*)d_in[18];
  const float* Wh   = (const float*)d_in[19];
  const float* bh   = (const float*)d_in[20];

  // --- carve workspace (recomputed every call; no cached state) ------------
  char* ws = (char*)d_ws;
  size_t off = 0;
  auto alloc = [&](size_t bytes) -> void* {
    off = (off + 255) & ~(size_t)255;
    void* p = ws + off;
    off += bytes;
    return p;
  };
  bf16_t* wq_t = (bf16_t*)alloc((size_t)LV * DV * DV  * 2);
  bf16_t* wk_t = (bf16_t*)alloc((size_t)LV * DV * DV  * 2);
  bf16_t* wv_t = (bf16_t*)alloc((size_t)LV * DV * DV  * 2);
  bf16_t* wo_t = (bf16_t*)alloc((size_t)LV * DV * DV  * 2);
  bf16_t* w1_t = (bf16_t*)alloc((size_t)LV * DV * D4V * 2);
  bf16_t* w2_t = (bf16_t*)alloc((size_t)LV * D4V * DV * 2);
  bf16_t* wh_t = (bf16_t*)alloc((size_t)DV * VV * 2);
  float*  x    = (float*) alloc((size_t)MV * DV * 4);
  bf16_t* hb   = (bf16_t*)alloc((size_t)MV * DV * 2);
  bf16_t* qb   = (bf16_t*)alloc((size_t)MV * DV * 2);
  bf16_t* kbuf = (bf16_t*)alloc((size_t)MV * DV * 2);
  bf16_t* vb   = (bf16_t*)alloc((size_t)MV * DV * 2);
  bf16_t* vt   = (bf16_t*)alloc((size_t)MV * DV * 2);
  bf16_t* obuf = (bf16_t*)alloc((size_t)MV * DV * 2);
  bf16_t* ub   = (bf16_t*)alloc((size_t)MV * D4V * 2);
  float*  rl   = (float*) alloc((size_t)MV * 4);

  const dim3 blk(256);

  // --- weights -> transposed bf16 ------------------------------------------
  for (int l = 0; l < LV; ++l) {
    size_t wo_off = (size_t)l * DV * DV;
    wconv_kernel<<<(DV * DV + 255) / 256, blk, 0, stream>>>(Wq + wo_off, wq_t + wo_off, DV, DV);
    wconv_kernel<<<(DV * DV + 255) / 256, blk, 0, stream>>>(Wk + wo_off, wk_t + wo_off, DV, DV);
    wconv_kernel<<<(DV * DV + 255) / 256, blk, 0, stream>>>(Wv + wo_off, wv_t + wo_off, DV, DV);
    wconv_kernel<<<(DV * DV + 255) / 256, blk, 0, stream>>>(Wo + wo_off, wo_t + wo_off, DV, DV);
    size_t w1_off = (size_t)l * DV * D4V;
    wconv_kernel<<<(DV * D4V + 255) / 256, blk, 0, stream>>>(W1 + w1_off, w1_t + w1_off, DV, D4V);
    wconv_kernel<<<(DV * D4V + 255) / 256, blk, 0, stream>>>(W2 + w1_off, w2_t + w1_off, D4V, DV);
  }
  wconv_kernel<<<((size_t)DV * VV + 255) / 256, blk, 0, stream>>>(Wh, wh_t, DV, VV);

  // --- embedding -----------------------------------------------------------
  embed_kernel<<<(MV * DV) / 256, blk, 0, stream>>>(idx, tok, pos, x);

  const float scale = 0.044194173824159216f;  // 512^-0.5 (full-D scaling, per ref)
  const dim3 gD(MV / 64, DV / 128);    // 64 x 4
  const dim3 g4(MV / 64, D4V / 128);   // 64 x 16
  const dim3 gV(MV / 64, VV / 128);    // 64 x 250

  for (int l = 0; l < LV; ++l) {
    size_t wsq = (size_t)l * DV * DV;
    size_t ws1 = (size_t)l * DV * D4V;
    // LN1 -> bf16
    ln_kernel<<<MV, 128, 0, stream>>>(x, ln1g + l * DV, ln1b + l * DV, hb, DV);
    // Q, K, V projections (bf16 out)
    gemm_kernel<<<gD, blk, 0, stream>>>(hb, wq_t + wsq, MV, DV, DV, nullptr, nullptr, nullptr, qb, 0);
    gemm_kernel<<<gD, blk, 0, stream>>>(hb, wk_t + wsq, MV, DV, DV, nullptr, nullptr, nullptr, kbuf, 0);
    gemm_kernel<<<gD, blk, 0, stream>>>(hb, wv_t + wsq, MV, DV, DV, nullptr, nullptr, nullptr, vb, 0);
    // V -> [B,H,hd,T]
    vtr_kernel<<<(MV * DV) / 256, blk, 0, stream>>>(vb, vt);
    // flash attention: 2048 waves / 8 per block
    attn_kernel<<<(BV * HV * (TV / 16)) / 8, blk, 0, stream>>>(qb, kbuf, vt, obuf, scale);
    // output projection + bias + residual -> x (fp32)
    gemm_kernel<<<gD, blk, 0, stream>>>(obuf, wo_t + wsq, MV, DV, DV, bo + l * DV, x, x, nullptr, 0);
    // LN2 -> bf16
    ln_kernel<<<MV, 128, 0, stream>>>(x, ln2g + l * DV, ln2b + l * DV, hb, DV);
    // MLP up + bias + relu (bf16 out), MLP down + bias + residual -> x
    gemm_kernel<<<g4, blk, 0, stream>>>(hb, w1_t + ws1, MV, D4V, DV, b1 + l * D4V, nullptr, nullptr, ub, 1);
    gemm_kernel<<<gD, blk, 0, stream>>>(ub, w2_t + ws1, MV, DV, D4V, b2 + l * DV, x, x, nullptr, 0);
  }

  // --- final LN, LM head, loss --------------------------------------------
  ln_kernel<<<MV, 128, 0, stream>>>(x, lnfg, lnfb, hb, DV);
  float* logits = (float*)d_out;
  gemm_kernel<<<gV, blk, 0, stream>>>(hb, wh_t, MV, VV, DV, bh, nullptr, logits, nullptr, 0);
  loss_row_kernel<<<MV, blk, 0, stream>>>(logits, tgt, rl);
  loss_reduce_kernel<<<1, blk, 0, stream>>>(rl, logits + (size_t)MV * VV);
}
